// TransformerLayer_23776938951167
// MI455X (gfx1250) — matrix operand order repaired and checked by reading, not compiled, not run on hardware
//
#include <hip/hip_runtime.h>

// ---------------- problem constants ----------------
#define B_  4
#define S_  2048
#define D_  1024
#define H_  16
#define HD_ 64
#define F_  4096
#define M_  (B_ * S_)        // 8192 rows of activations

typedef __attribute__((ext_vector_type(16))) __bf16 v16bf;
typedef __attribute__((ext_vector_type(8)))  float  v8f;

union U16 { v16bf v; unsigned short u[16]; uint4 q[2]; };

static __device__ __forceinline__ unsigned short f2bf(float f) {
  union { float f; unsigned u; } a; a.f = f;
  unsigned u = a.u;
  unsigned r = u + 0x7FFFu + ((u >> 16) & 1u);   // round-to-nearest-even
  return (unsigned short)(r >> 16);
}
static __device__ __forceinline__ float bf2f(unsigned short h) {
  union { unsigned u; float f; } a; a.u = ((unsigned)h) << 16; return a.f;
}
static __device__ __forceinline__ float gelu_tanh(float x) {
  return 0.5f * x * (1.0f + tanhf(0.79788456080286535588f * (x + 0.044715f * x * x * x)));
}

// ---------------- fp32 -> bf16 weight convert + B-fragment swizzle ----------------
// Wp layout: [N/16][K/32][lane(32)][16], so a lane's whole WMMA B fragment is
// 32 contiguous bytes (two global_load_b128), coalesced across the wave.
// Fragment semantics: lane L holds B[kblk*32 + ((j&8)<<1) + ((L<16)?0:8) + (j&7)][nblk*16 + (L&15)].
__global__ void convert_swizzle_b(const float* __restrict__ W,
                                  unsigned short* __restrict__ Wp,
                                  int Ni, int Ki) {
  const int idx = blockIdx.x * blockDim.x + threadIdx.x;
  if (idx >= Ni * Ki) return;
  const int j    = idx & 15;
  const int lane = (idx >> 4) & 31;
  const int rest = idx >> 9;
  const int kblocks = Ki >> 5;
  const int kb = rest % kblocks;
  const int nb = rest / kblocks;
  const int k = kb * 32 + ((j & 8) << 1) + ((lane < 16) ? 0 : 8) + (j & 7);
  const int n = nb * 16 + (lane & 15);
  Wp[idx] = f2bf(W[(size_t)k * Ni + n]);
}

// ---------------- RMSNorm (one block per row) ----------------
__global__ __launch_bounds__(256) void rmsnorm_bf16(const float* __restrict__ x,
                                                    const float* __restrict__ scale,
                                                    unsigned short* __restrict__ out) {
  __shared__ float red[256];
  const int row = blockIdx.x;
  const float* xr = x + (size_t)row * D_;
  float vals[4];
  float s = 0.0f;
  #pragma unroll
  for (int i = 0; i < 4; ++i) {
    vals[i] = xr[threadIdx.x + i * 256];
    s += vals[i] * vals[i];
  }
  red[threadIdx.x] = s;
  __syncthreads();
  for (int off = 128; off > 0; off >>= 1) {
    if (threadIdx.x < off) red[threadIdx.x] += red[threadIdx.x + off];
    __syncthreads();
  }
  const float inv = rsqrtf(red[0] * (1.0f / D_) + 1e-6f);
  #pragma unroll
  for (int i = 0; i < 4; ++i) {
    const int c = threadIdx.x + i * 256;
    out[(size_t)row * D_ + c] = f2bf(vals[i] * inv * scale[c]);
  }
}

// ---------------- RoPE (in-place on bf16 Q or K), optional 1/sqrt(hd) ----------------
__global__ void rope_kernel(unsigned short* qk, const int* __restrict__ pos,
                            float outscale) {
  const int idx = blockIdx.x * blockDim.x + threadIdx.x;   // B*S*H*32 threads
  const int i = idx & 31;
  const int h = (idx >> 5) & (H_ - 1);
  const int s = (idx >> 9) & (S_ - 1);
  const int b = idx >> 20;
  const size_t base = ((size_t)(b * S_ + s)) * D_ + h * HD_ + i;
  const float first  = bf2f(qk[base]);
  const float second = bf2f(qk[base + 32]);
  const float p = (float)pos[b * S_ + s];
  const float ang = p * __powf(10000.0f, -(float)i * (1.0f / 32.0f));
  float sn, cs;
  __sincosf(ang, &sn, &cs);
  qk[base]      = f2bf((first * cs - second * sn) * outscale);
  qk[base + 32] = f2bf((second * cs + first * sn) * outscale);
}

// ---------------- WMMA GEMM ----------------
// C[M,N] = A[M,K] * Bp[K,N-preswizzled]; block tile 128x128, K-step 32,
// 8 waves (2x4), wave tile 64x32. A staged via LDS (ds_load_b128 frags),
// B fragments loaded straight from pre-swizzled global memory (b128, L2-resident).
// EPI: 0 = bf16 row-major out; 1 = bf16 out transposed to vT[b,h,d,s]; 2 = fp32 out + residual.
template <int EPI>
__global__ __launch_bounds__(256) void gemm_wmma(
    const unsigned short* __restrict__ A, const unsigned short* __restrict__ Bp,
    const float* __restrict__ R, void* __restrict__ Cv, int Ni, int Ki) {
  __shared__ __align__(16) unsigned short As[128][40];
  const int tid = threadIdx.x;
  const int lane = tid & 31;
  const int wave = tid >> 5;
  const int waveM = wave >> 2, waveN = wave & 3;
  const int m0 = blockIdx.y * 128, n0 = blockIdx.x * 128;
  const int rowA = tid >> 1, halfA = tid & 1;
  const int r16 = lane & 15;
  const int abase = (lane < 16) ? 0 : 8;
  const int kblocks = Ki >> 5;

  const v8f vzero = {0.f,0.f,0.f,0.f,0.f,0.f,0.f,0.f};
  v8f acc[4][2];
  for (int i = 0; i < 4; ++i) for (int j = 0; j < 2; ++j) acc[i][j] = vzero;

  const size_t aRowBase = (size_t)(m0 + rowA) * Ki + halfA * 16;

  for (int kc = 0; kc < Ki; kc += 32) {
    __syncthreads();
    {
      const uint4* sa = reinterpret_cast<const uint4*>(A + aRowBase + kc);
      uint4* da = reinterpret_cast<uint4*>(&As[rowA][halfA * 16]);
      da[0] = sa[0]; da[1] = sa[1];
      __builtin_prefetch(A + aRowBase + kc + 32, 0, 3);   // global_prefetch_b8
    }
    __syncthreads();
    U16 af[4], bfr[2];
    #pragma unroll
    for (int mi = 0; mi < 4; ++mi) {
      const int row = waveM * 64 + mi * 16 + r16;
      #pragma unroll
      for (int j = 0; j < 8; ++j) {
        af[mi].u[j]     = As[row][abase + j];
        af[mi].u[8 + j] = As[row][16 + abase + j];
      }
    }
    #pragma unroll
    for (int ni = 0; ni < 2; ++ni) {
      const size_t fb = ((((size_t)(n0 >> 4) + waveN * 2 + ni) * kblocks + (kc >> 5)) * 32 + lane) * 16;
      const uint4* sb = reinterpret_cast<const uint4*>(Bp + fb);
      bfr[ni].q[0] = sb[0];
      bfr[ni].q[1] = sb[1];
    }
    #pragma unroll
    for (int mi = 0; mi < 4; ++mi)
      #pragma unroll
      for (int ni = 0; ni < 2; ++ni)
        acc[mi][ni] = __builtin_amdgcn_wmma_f32_16x16x32_bf16(
            false, af[mi].v, false, bfr[ni].v, (short)0, acc[mi][ni], false, false);
  }
  const int rowoff = (lane < 16) ? 0 : 8;
  for (int mi = 0; mi < 4; ++mi)
    for (int ni = 0; ni < 2; ++ni)
      for (int r = 0; r < 8; ++r) {
        const int row = m0 + waveM * 64 + mi * 16 + rowoff + r;
        const int col = n0 + waveN * 32 + ni * 16 + r16;
        const float v = acc[mi][ni][r];
        if (EPI == 0) {
          ((unsigned short*)Cv)[(size_t)row * Ni + col] = f2bf(v);
        } else if (EPI == 1) {
          // vT[b][h][d][s]  (row = b*S+s over activations, col = h*64+d)
          const int b = row >> 11, s = row & (S_ - 1);
          const int h = col >> 6,  d = col & 63;
          ((unsigned short*)Cv)[(((size_t)b * H_ + h) * HD_ + d) * S_ + s] = f2bf(v);
        } else {
          const size_t idx = (size_t)row * Ni + col;
          ((float*)Cv)[idx] = v + R[idx];
        }
      }
}

// ---------------- fused gate/up GEMM: out = gelu(A*Wg) * (A*Wu), bf16 out ----------------
// block tile 128x64, wave tile 64x16, two accumulator sets; B frags from pre-swizzled global
__global__ __launch_bounds__(256) void gemm_gateup(
    const unsigned short* __restrict__ A, const unsigned short* __restrict__ Wg,
    const unsigned short* __restrict__ Wu, unsigned short* __restrict__ Out,
    int Ni, int Ki) {
  __shared__ __align__(16) unsigned short As[128][40];
  const int tid = threadIdx.x;
  const int lane = tid & 31;
  const int wave = tid >> 5;
  const int waveM = wave >> 2, waveN = wave & 3;
  const int m0 = blockIdx.y * 128, n0 = blockIdx.x * 64;
  const int rowA = tid >> 1, halfA = tid & 1;
  const int r16 = lane & 15;
  const int abase = (lane < 16) ? 0 : 8;
  const int kblocks = Ki >> 5;

  const v8f vzero = {0.f,0.f,0.f,0.f,0.f,0.f,0.f,0.f};
  v8f accG[4], accU[4];
  for (int i = 0; i < 4; ++i) { accG[i] = vzero; accU[i] = vzero; }

  const size_t aRowBase = (size_t)(m0 + rowA) * Ki + halfA * 16;

  for (int kc = 0; kc < Ki; kc += 32) {
    __syncthreads();
    {
      const uint4* sa = reinterpret_cast<const uint4*>(A + aRowBase + kc);
      uint4* da = reinterpret_cast<uint4*>(&As[rowA][halfA * 16]);
      da[0] = sa[0]; da[1] = sa[1];
      __builtin_prefetch(A + aRowBase + kc + 32, 0, 3);
    }
    __syncthreads();
    U16 af[4], bgf, buf;
    #pragma unroll
    for (int mi = 0; mi < 4; ++mi) {
      const int row = waveM * 64 + mi * 16 + r16;
      #pragma unroll
      for (int j = 0; j < 8; ++j) {
        af[mi].u[j]     = As[row][abase + j];
        af[mi].u[8 + j] = As[row][16 + abase + j];
      }
    }
    {
      const size_t fb = ((((size_t)(n0 >> 4) + waveN) * kblocks + (kc >> 5)) * 32 + lane) * 16;
      const uint4* sg = reinterpret_cast<const uint4*>(Wg + fb);
      bgf.q[0] = sg[0]; bgf.q[1] = sg[1];
      const uint4* su = reinterpret_cast<const uint4*>(Wu + fb);
      buf.q[0] = su[0]; buf.q[1] = su[1];
    }
    #pragma unroll
    for (int mi = 0; mi < 4; ++mi) {
      accG[mi] = __builtin_amdgcn_wmma_f32_16x16x32_bf16(
          false, af[mi].v, false, bgf.v, (short)0, accG[mi], false, false);
      accU[mi] = __builtin_amdgcn_wmma_f32_16x16x32_bf16(
          false, af[mi].v, false, buf.v, (short)0, accU[mi], false, false);
    }
  }
  const int rowoff = (lane < 16) ? 0 : 8;
  for (int mi = 0; mi < 4; ++mi)
    for (int r = 0; r < 8; ++r) {
      const int row = m0 + waveM * 64 + mi * 16 + rowoff + r;
      const int col = n0 + waveN * 16 + r16;
      Out[(size_t)row * Ni + col] = f2bf(gelu_tanh(accG[mi][r]) * accU[mi][r]);
    }
}

// ---------------- flash attention: one wave per 16-query tile, 32-key chunks ----------------
// K tiles are [key][d] in LDS (K-frag reads contiguous); V comes in pre-transposed
// global layout vT[b,h,d,s], staged as VsT[d][key] (V-frag reads contiguous).
__global__ __launch_bounds__(256) void attn_kernel(
    const unsigned short* __restrict__ Q, const unsigned short* __restrict__ K,
    const unsigned short* __restrict__ Vt, unsigned short* __restrict__ Ctx) {
  __shared__ __align__(16) unsigned short Ks[32][72];
  __shared__ __align__(16) unsigned short VsT[64][40];
  __shared__ __align__(16) unsigned short Pst[8][16][40];
  const int tid = threadIdx.x;
  const int lane = tid & 31;
  const int w = tid >> 5;
  const int b = blockIdx.y / H_;
  const int h = blockIdx.y % H_;
  const int q0 = blockIdx.x * 128 + w * 16;
  const int r16 = lane & 15;
  const int abase = (lane < 16) ? 0 : 8;
  const int koff  = (lane < 16) ? 0 : 16;
  const int rowoff = (lane < 16) ? 0 : 8;
  const int rowK = tid >> 3, segK = tid & 7;   // 32 rows x 8 segs of 8
  const int rowV = tid >> 2, segV = tid & 3;   // 64 rows x 4 segs of 8

  // Q fragments (A operand), two K-dim chunks of 32 over head-dim 64
  U16 qa[2];
  #pragma unroll
  for (int c = 0; c < 2; ++c) {
    const size_t base = ((size_t)(b * S_ + q0 + r16)) * D_ + h * HD_ + c * 32;
    #pragma unroll
    for (int j = 0; j < 8; ++j) {
      qa[c].u[j]     = Q[base + abase + j];
      qa[c].u[8 + j] = Q[base + 16 + abase + j];
    }
  }

  const v8f vzero = {0.f,0.f,0.f,0.f,0.f,0.f,0.f,0.f};
  v8f accO[4];
  for (int i = 0; i < 4; ++i) accO[i] = vzero;
  float mrow[8], lrow[8];
  #pragma unroll
  for (int r = 0; r < 8; ++r) { mrow[r] = -3.0e38f; lrow[r] = 0.0f; }

  const int nchunks = blockIdx.x * 4 + 4;   // keys up to block's max query
  for (int kc = 0; kc < nchunks; ++kc) {
    const int key0 = kc * 32;
    __syncthreads();
    {
      const size_t srcK = ((size_t)(b * S_ + key0 + rowK)) * D_ + h * HD_ + segK * 8;
      reinterpret_cast<uint4*>(&Ks[rowK][segK * 8])[0] =
          reinterpret_cast<const uint4*>(K + srcK)[0];
      const size_t srcV = (((size_t)b * H_ + h) * HD_ + rowV) * S_ + key0 + segV * 8;
      reinterpret_cast<uint4*>(&VsT[rowV][segV * 8])[0] =
          reinterpret_cast<const uint4*>(Vt + srcV)[0];
    }
    __syncthreads();
    if (key0 <= q0 + 15) {   // wave-uniform causal skip
      // ---- scores: two 16x16 tiles over 32 keys, K-dim = 64 (2 WMMAs each)
      v8f sc[2] = {vzero, vzero};
      #pragma unroll
      for (int t = 0; t < 2; ++t)
        #pragma unroll
        for (int c = 0; c < 2; ++c) {
          U16 bk;
          #pragma unroll
          for (int j = 0; j < 16; ++j) bk.u[j] = Ks[t * 16 + r16][c * 32 + abase + ((j & 8) << 1) + (j & 7)];
          sc[t] = __builtin_amdgcn_wmma_f32_16x16x32_bf16(
              false, qa[c].v, false, bk.v, (short)0, sc[t], false, false);
        }
      // ---- online softmax over the 32-key chunk (rows live in 16-lane halves)
      #pragma unroll
      for (int r = 0; r < 8; ++r) {
        const int qrow = q0 + rowoff + r;
        float s0 = sc[0][r]; if (key0 + r16 > qrow)      s0 = -3.0e38f;
        float s1 = sc[1][r]; if (key0 + 16 + r16 > qrow) s1 = -3.0e38f;
        float mx = fmaxf(s0, s1);
        mx = fmaxf(mx, __shfl_xor(mx, 1, 32));
        mx = fmaxf(mx, __shfl_xor(mx, 2, 32));
        mx = fmaxf(mx, __shfl_xor(mx, 4, 32));
        mx = fmaxf(mx, __shfl_xor(mx, 8, 32));
        const float mnew = fmaxf(mrow[r], mx);
        const float corr = __expf(mrow[r] - mnew);
        mrow[r] = mnew;
        const float p0 = __expf(s0 - mnew);
        const float p1 = __expf(s1 - mnew);
        float ps = p0 + p1;
        ps += __shfl_xor(ps, 1, 32);
        ps += __shfl_xor(ps, 2, 32);
        ps += __shfl_xor(ps, 4, 32);
        ps += __shfl_xor(ps, 8, 32);
        lrow[r] = lrow[r] * corr + ps;
        #pragma unroll
        for (int nt = 0; nt < 4; ++nt) accO[nt][r] = accO[nt][r] * corr;
        Pst[w][rowoff + r][r16]      = f2bf(p0);
        Pst[w][rowoff + r][16 + r16] = f2bf(p1);
      }
      asm volatile("s_wait_dscnt 0x0" ::: "memory");  // intra-wave LDS store->load order
      // ---- P (A operand) and PV accumulation
      U16 pa;
      #pragma unroll
      for (int j = 0; j < 8; ++j) {
        pa.u[j]     = Pst[w][r16][abase + j];
        pa.u[8 + j] = Pst[w][r16][16 + abase + j];
      }
      #pragma unroll
      for (int nt = 0; nt < 4; ++nt) {
        U16 bv;
        #pragma unroll
        for (int j = 0; j < 16; ++j) bv.u[j] = VsT[nt * 16 + r16][abase + ((j & 8) << 1) + (j & 7)];
        accO[nt] = __builtin_amdgcn_wmma_f32_16x16x32_bf16(
            false, pa.v, false, bv.v, (short)0, accO[nt], false, false);
      }
    }
  }
  // ---- normalize and store ctx (layout [s, h*64+d])
  #pragma unroll
  for (int r = 0; r < 8; ++r) {
    const float inv = 1.0f / lrow[r];
    const int qrow = q0 + rowoff + r;
    #pragma unroll
    for (int nt = 0; nt < 4; ++nt) {
      const int d = nt * 16 + r16;
      Ctx[((size_t)(b * S_ + qrow)) * D_ + h * HD_ + d] = f2bf(accO[nt][r] * inv);
    }
  }
}

// ---------------- workspace layout (bytes) ----------------
static const size_t OFF_WQ  = 0;                       // 1024*1024 bf16 = 2 MB (pre-swizzled)
static const size_t OFF_WK  = OFF_WQ  + 2097152;
static const size_t OFF_WV  = OFF_WK  + 2097152;
static const size_t OFF_WO  = OFF_WV  + 2097152;
static const size_t OFF_WG  = OFF_WO  + 2097152;       // 1024*4096 bf16 = 8 MB
static const size_t OFF_WU  = OFF_WG  + 8388608;
static const size_t OFF_WD  = OFF_WU  + 8388608;       // 4096*1024 bf16 = 8 MB
static const size_t OFF_H1  = OFF_WD  + 8388608;       // 8192*1024 bf16 = 16 MB
static const size_t OFF_Q   = OFF_H1  + 16777216;
static const size_t OFF_K   = OFF_Q   + 16777216;
static const size_t OFF_V   = OFF_K   + 16777216;      // stored as vT[b,h,d,s]
static const size_t OFF_CTX = OFF_V   + 16777216;
static const size_t OFF_X1  = OFF_CTX + 16777216;      // fp32 residual, 32 MB
static const size_t OFF_H2  = OFF_X1  + 33554432;
static const size_t OFF_ACT = OFF_H2  + 16777216;      // 8192*4096 bf16 = 64 MB
// total ~224 MB

extern "C" void kernel_launch(void* const* d_in, const int* in_sizes, int n_in,
                              void* d_out, int out_size, void* d_ws, size_t ws_size,
                              hipStream_t stream) {
  (void)in_sizes; (void)n_in; (void)out_size; (void)ws_size;
  const float* x     = (const float*)d_in[0];
  const int*   pos   = (const int*)d_in[1];
  // d_in[2]: mask — causality is applied analytically
  const float* wq    = (const float*)d_in[3];
  const float* wk    = (const float*)d_in[4];
  const float* wv    = (const float*)d_in[5];
  const float* wo    = (const float*)d_in[6];
  const float* wg    = (const float*)d_in[7];
  const float* wu    = (const float*)d_in[8];
  const float* wd    = (const float*)d_in[9];
  const float* norm1 = (const float*)d_in[10];
  const float* norm2 = (const float*)d_in[11];
  float* out = (float*)d_out;

  char* ws = (char*)d_ws;
  unsigned short* wq_b  = (unsigned short*)(ws + OFF_WQ);
  unsigned short* wk_b  = (unsigned short*)(ws + OFF_WK);
  unsigned short* wv_b  = (unsigned short*)(ws + OFF_WV);
  unsigned short* wo_b  = (unsigned short*)(ws + OFF_WO);
  unsigned short* wg_b  = (unsigned short*)(ws + OFF_WG);
  unsigned short* wu_b  = (unsigned short*)(ws + OFF_WU);
  unsigned short* wd_b  = (unsigned short*)(ws + OFF_WD);
  unsigned short* h1_b  = (unsigned short*)(ws + OFF_H1);
  unsigned short* q_b   = (unsigned short*)(ws + OFF_Q);
  unsigned short* k_b   = (unsigned short*)(ws + OFF_K);
  unsigned short* vT_b  = (unsigned short*)(ws + OFF_V);
  unsigned short* ctx_b = (unsigned short*)(ws + OFF_CTX);
  float*          x1    = (float*)(ws + OFF_X1);
  unsigned short* h2_b  = (unsigned short*)(ws + OFF_H2);
  unsigned short* act_b = (unsigned short*)(ws + OFF_ACT);

  // 1) weights: fp32 -> bf16 + B-fragment swizzle
  const int nDN = D_ * D_;         // 1,048,576
  const int nDF = D_ * F_;         // 4,194,304
  convert_swizzle_b<<<(nDN + 255) / 256, 256, 0, stream>>>(wq, wq_b, D_, D_);
  convert_swizzle_b<<<(nDN + 255) / 256, 256, 0, stream>>>(wk, wk_b, D_, D_);
  convert_swizzle_b<<<(nDN + 255) / 256, 256, 0, stream>>>(wv, wv_b, D_, D_);
  convert_swizzle_b<<<(nDN + 255) / 256, 256, 0, stream>>>(wo, wo_b, D_, D_);
  convert_swizzle_b<<<(nDF + 255) / 256, 256, 0, stream>>>(wg, wg_b, F_, D_);
  convert_swizzle_b<<<(nDF + 255) / 256, 256, 0, stream>>>(wu, wu_b, F_, D_);
  convert_swizzle_b<<<(nDF + 255) / 256, 256, 0, stream>>>(wd, wd_b, D_, F_);

  // 2) rmsnorm1
  rmsnorm_bf16<<<M_, 256, 0, stream>>>(x, norm1, h1_b);

  // 3) Q,K,V projections: [8192,1024] x [1024,1024]; V written transposed
  dim3 gProj(D_ / 128, M_ / 128);
  gemm_wmma<0><<<gProj, 256, 0, stream>>>(h1_b, wq_b, nullptr, q_b,  D_, D_);
  gemm_wmma<0><<<gProj, 256, 0, stream>>>(h1_b, wk_b, nullptr, k_b,  D_, D_);
  gemm_wmma<1><<<gProj, 256, 0, stream>>>(h1_b, wv_b, nullptr, vT_b, D_, D_);

  // 4) RoPE (in place); q also gets 1/sqrt(64)
  const int nRope = B_ * S_ * H_ * 32;
  rope_kernel<<<nRope / 256, 256, 0, stream>>>(q_b, pos, 0.125f);
  rope_kernel<<<nRope / 256, 256, 0, stream>>>(k_b, pos, 1.0f);

  // 5) attention
  dim3 gAttn(S_ / 128, B_ * H_);
  attn_kernel<<<gAttn, 256, 0, stream>>>(q_b, k_b, vT_b, ctx_b);

  // 6) output projection + residual -> x1 (fp32)
  gemm_wmma<2><<<gProj, 256, 0, stream>>>(ctx_b, wo_b, x, x1, D_, D_);

  // 7) rmsnorm2
  rmsnorm_bf16<<<M_, 256, 0, stream>>>(x1, norm2, h2_b);

  // 8) fused gate/up + gelu-mul: [8192,1024] x [1024,4096] (x2)
  dim3 gGU(F_ / 64, M_ / 128);
  gemm_gateup<<<gGU, 256, 0, stream>>>(h2_b, wg_b, wu_b, act_b, F_, D_);

  // 9) down projection + residual -> d_out (fp32): [8192,4096] x [4096,1024]
  gemm_wmma<2><<<gProj, 256, 0, stream>>>(act_b, wd_b, x1, out, D_, F_);
}
